// minGRU_Reinformer_72318659330182
// MI455X (gfx1250) — matrix operand order repaired and checked by reading, not compiled, not run on hardware
//
#include <hip/hip_runtime.h>

typedef __bf16 bf16;
typedef __attribute__((ext_vector_type(16))) __bf16 v16bf;
typedef __attribute__((ext_vector_type(8)))  __bf16 v8bf;
typedef __attribute__((ext_vector_type(8)))  float  v8f;

#define BB 16
#define TT 512
#define HH 512
#define LL 1536              // 3*T
#define BL (BB*LL)           // 24576 rows
#define NB 6
#define H2 1024              // 2*H
#define SDIM 17
#define ADIM 6

// ---- workspace layout (bytes) ----
#define WT_BYTES   ((size_t)NB * H2 * HH * 2)          // 6,291,456  : bf16 W^T [6][1024][512]
#define HBF_BYTES  ((size_t)BL * HH * 2)               // 25,165,824 : bf16 activations [BL][512]
// hg fp32 [BL][1024] = 100,663,296 follows

// ---- CDNA5 async global->LDS path (gated; falls back to register staging) ----
#if defined(__has_builtin)
#if __has_builtin(__builtin_amdgcn_global_load_async_to_lds_b128) && \
    __has_builtin(__builtin_amdgcn_s_wait_asynccnt)
#define USE_ASYNC_LDS 1
#endif
#endif
#ifndef USE_ASYNC_LDS
#define USE_ASYNC_LDS 0
#endif

#if USE_ASYNC_LDS
// builtin signature (from compiler diagnostic): (v4i AS1*, v4i AS3*, imm i32, imm i32)
typedef int v4i __attribute__((vector_size(16)));
typedef __attribute__((address_space(1))) v4i g_v4i;
typedef __attribute__((address_space(3))) v4i l_v4i;
// generic LDS pointers carry the AS3 offset in their low 32 bits; generic
// global pointers are the 64-bit VA. Integer round-trip avoids addrspacecast
// legality questions in HIP mode.
__device__ __forceinline__ void async_b128(const void* g, void* l) {
    __builtin_amdgcn_global_load_async_to_lds_b128(
        (g_v4i*)(unsigned long long)g,
        (l_v4i*)(unsigned int)(unsigned long long)l, 0, 0);
}
#endif

// ---------------------------------------------------------------------------
// 0. Convert + transpose GRU weights: f32 [6][512][1024] -> bf16 [6][1024][512]
// ---------------------------------------------------------------------------
__global__ void wt_convert_kernel(const float* __restrict__ gw, bf16* __restrict__ wt) {
    __shared__ float tile[32][33];
    const int layer = blockIdx.z;
    const int kb = blockIdx.y * 32;      // k tile (0..511)
    const int nb = blockIdx.x * 32;      // n tile (0..1023)
    const int tx = threadIdx.x, ty = threadIdx.y;   // (32, 8)
#pragma unroll
    for (int i = 0; i < 32; i += 8)
        tile[ty + i][tx] = gw[(size_t)layer * HH * H2 + (size_t)(kb + ty + i) * H2 + nb + tx];
    __syncthreads();
#pragma unroll
    for (int i = 0; i < 32; i += 8)
        wt[(size_t)layer * H2 * HH + (size_t)(nb + ty + i) * HH + kb + tx] = (bf16)tile[tx][ty + i];
}

// ---------------------------------------------------------------------------
// 1. Embedding + input projection + LayerNorm -> bf16 h0 [BL][512]
//    token order per t: (state, rtg, action)
// ---------------------------------------------------------------------------
__global__ void embed_ln_kernel(const int* __restrict__ timesteps,
                                const float* __restrict__ states,
                                const float* __restrict__ actions,
                                const float* __restrict__ rtg,
                                const float* __restrict__ emb_t,
                                const float* __restrict__ Ws, const float* __restrict__ bs,
                                const float* __restrict__ Wr, const float* __restrict__ br,
                                const float* __restrict__ Wa, const float* __restrict__ ba,
                                const float* __restrict__ ln_g, const float* __restrict__ ln_b,
                                bf16* __restrict__ hbf) {
    const int row = blockIdx.x;          // 0..BL-1
    const int b = row / LL, l = row - b * LL;
    const int t = l / 3, j = l - 3 * t;
    const int ts = timesteps[b * TT + t];
    const float* et = emb_t + (size_t)ts * HH;
    const int tid = threadIdx.x;         // 128 threads

    const float* W; const float* bias; const float* x; int K;
    if (j == 0)      { W = Ws; bias = bs; K = SDIM; x = states + (size_t)(b * TT + t) * SDIM; }
    else if (j == 1) { W = Wr; bias = br; K = 1;    x = rtg + (b * TT + t); }
    else             { W = Wa; bias = ba; K = ADIM; x = actions + (size_t)(b * TT + t) * ADIM; }

    float xr[SDIM];
    for (int k = 0; k < K; k++) xr[k] = x[k];

    float y[4], s1 = 0.f, s2 = 0.f;
#pragma unroll
    for (int i = 0; i < 4; i++) {
        const int c = tid + i * 128;
        float acc = bias[c] + et[c];
        for (int k = 0; k < K; k++) acc += xr[k] * W[k * HH + c];
        y[i] = acc; s1 += acc; s2 += acc * acc;
    }
    __shared__ float r1[128], r2[128];
    r1[tid] = s1; r2[tid] = s2;
    __syncthreads();
    for (int off = 64; off > 0; off >>= 1) {
        if (tid < off) { r1[tid] += r1[tid + off]; r2[tid] += r2[tid + off]; }
        __syncthreads();
    }
    const float mean = r1[0] * (1.f / HH);
    const float var  = r2[0] * (1.f / HH) - mean * mean;
    const float rstd = rsqrtf(var + 1e-5f);
#pragma unroll
    for (int i = 0; i < 4; i++) {
        const int c = tid + i * 128;
        hbf[(size_t)row * HH + c] = (bf16)((y[i] - mean) * rstd * ln_g[c] + ln_b[c]);
    }
}

// ---------------------------------------------------------------------------
// 2. WMMA GEMM: hg[BL][1024] = Hbf[BL][512] @ Wt^T + bias
//    block tile 128x128, 8 waves (2x4), wave tile 64x32, K step 32.
//    Async path: double-buffered LDS, next K-tile in flight via ASYNCcnt.
// ---------------------------------------------------------------------------
#define LDT 40   // padded LDS row stride (bf16 elems): keeps 16B alignment, breaks bank cycles

__global__ __launch_bounds__(256) void gemm_kernel(const bf16* __restrict__ Ah,
                                                   const bf16* __restrict__ Bw,
                                                   const float* __restrict__ bias2h,
                                                   float* __restrict__ hg) {
    const int tid  = threadIdx.x;
    const int lane = tid & 31;
    const int w    = tid >> 5;
    const int wm   = w & 1;              // 2 waves along M
    const int wn   = w >> 1;             // 4 waves along N
    const int m0   = blockIdx.y * 128;
    const int n0   = blockIdx.x * 128;

    __shared__ bf16 Als[2][128 * LDT];
    __shared__ bf16 Bls[2][128 * LDT];

    v8f acc[4][2] = {};

    const int srow  = tid >> 1;          // 0..127 : tile row (A) / tile col (B)
    const int shalf = tid & 1;           // which 16-elem half of the K=32 slab
    const int soff  = srow * LDT + shalf * 16;
    const bf16* gA = Ah + (size_t)(m0 + srow) * HH + shalf * 16;
    const bf16* gB = Bw + (size_t)(n0 + srow) * HH + shalf * 16;

    const int akb = (lane < 16) ? 0 : 8;     // A fragment K base (per ISA lane layout)
    const int bkb = (lane < 16) ? 0 : 8;     // B fragment K base
    const int l15 = lane & 15;

#if USE_ASYNC_LDS
    // prologue: tiles 0 and 1 in flight
    async_b128(gA,      &Als[0][soff]);   async_b128(gA + 8,  &Als[0][soff + 8]);
    async_b128(gB,      &Bls[0][soff]);   async_b128(gB + 8,  &Bls[0][soff + 8]);
    async_b128(gA + 32, &Als[1][soff]);   async_b128(gA + 40, &Als[1][soff + 8]);
    async_b128(gB + 32, &Bls[1][soff]);   async_b128(gB + 40, &Bls[1][soff + 8]);
    gA += 64; gB += 64;
#endif

    for (int kt = 0; kt < 16; kt++) {
        const int cur = USE_ASYNC_LDS ? (kt & 1) : 0;
#if USE_ASYNC_LDS
        if (kt < 15) __builtin_amdgcn_s_wait_asynccnt(4);   // tile kt done, kt+1 in flight
        else         __builtin_amdgcn_s_wait_asynccnt(0);
        __syncthreads();
#else
        v8bf a0 = *(const v8bf*)(gA);
        v8bf a1 = *(const v8bf*)(gA + 8);
        v8bf b0 = *(const v8bf*)(gB);
        v8bf b1 = *(const v8bf*)(gB + 8);
        if (kt < 15) {
            __builtin_prefetch(gA + 32, 0, 3);
            __builtin_prefetch(gB + 32, 0, 3);
        }
        gA += 32; gB += 32;
        __syncthreads();
        *(v8bf*)&Als[0][soff]     = a0;
        *(v8bf*)&Als[0][soff + 8] = a1;
        *(v8bf*)&Bls[0][soff]     = b0;
        *(v8bf*)&Bls[0][soff + 8] = b1;
        __syncthreads();
#endif
        const bf16* As = Als[cur];
        const bf16* Bs = Bls[cur];

        v16bf af[4], bfr[2];
#pragma unroll
        for (int mi = 0; mi < 4; mi++) {
            const int r = wm * 64 + mi * 16 + l15;
            v8bf lo = *(const v8bf*)&As[r * LDT + akb];        // K = akb..akb+7
            v8bf hi = *(const v8bf*)&As[r * LDT + akb + 16];   // K = akb+16..akb+23
#pragma unroll
            for (int e = 0; e < 8; e++) { af[mi][e] = lo[e]; af[mi][e + 8] = hi[e]; }
        }
#pragma unroll
        for (int ni = 0; ni < 2; ni++) {
            const int c = wn * 32 + ni * 16 + l15;
            v8bf lo = *(const v8bf*)&Bs[c * LDT + bkb];        // K = bkb..bkb+7
            v8bf hi = *(const v8bf*)&Bs[c * LDT + bkb + 16];   // K = bkb+16..bkb+23
#pragma unroll
            for (int e = 0; e < 8; e++) { bfr[ni][e] = lo[e]; bfr[ni][e + 8] = hi[e]; }
        }
#pragma unroll
        for (int mi = 0; mi < 4; mi++)
#pragma unroll
            for (int ni = 0; ni < 2; ni++)
                acc[mi][ni] = __builtin_amdgcn_wmma_f32_16x16x32_bf16(
                    false, af[mi], false, bfr[ni], (short)0, acc[mi][ni], false, false);

#if USE_ASYNC_LDS
        __syncthreads();                       // everyone done reading buf[cur]
        if (kt + 2 < 16) {                     // refill it with tile kt+2
            async_b128(gA,     &Als[cur][soff]);
            async_b128(gA + 8, &Als[cur][soff + 8]);
            async_b128(gB,     &Bls[cur][soff]);
            async_b128(gB + 8, &Bls[cur][soff + 8]);
            gA += 32; gB += 32;
        }
#endif
    }

    // epilogue: C/D layout -> lane<16: row=r,col=lane ; lane>=16: row=r+8,col=lane-16
    const int rowoff = (lane < 16) ? 0 : 8;
#pragma unroll
    for (int ni = 0; ni < 2; ni++) {
        const int gcol = n0 + wn * 32 + ni * 16 + l15;
        const float bb = bias2h[gcol];
#pragma unroll
        for (int mi = 0; mi < 4; mi++) {
            const int gr0 = m0 + wm * 64 + mi * 16 + rowoff;
#pragma unroll
            for (int r = 0; r < 8; r++)
                hg[(size_t)(gr0 + r) * H2 + gcol] = acc[mi][ni][r] + bb;
        }
    }
}

// ---------------------------------------------------------------------------
// 3. minGRU gate + chunked time scan. hg is L2-resident (100MB < 192MB), so
//    pass B recomputes a,u from hg instead of spilling them.
//    grid: B*16 blocks, block (32 h-lanes, 8 time-chunks of 192)
// ---------------------------------------------------------------------------
__device__ __forceinline__ void gate_au(float ht, float gt, float& a, float& u) {
    const float z = 1.f / (1.f + __expf(-gt));
    a = 1.f - z;
    const float g = (ht >= 0.f) ? (ht + 0.5f) : 1.f / (1.f + __expf(-ht));
    u = z * g;
}

__global__ void scan_kernel(const float* __restrict__ hg, bf16* __restrict__ hbf) {
    const int b     = blockIdx.x >> 4;
    const int htile = blockIdx.x & 15;
    const int h     = htile * 32 + threadIdx.x;
    const int chunk = threadIdx.y;            // 0..7
    const int CH    = LL / 8;                 // 192
    const int l0    = chunk * CH;
    const size_t rowbase = (size_t)b * LL;

    __shared__ float Ag[8][32], Ug[8][32], Pu[8][32];

    float A = 1.f, U = 0.f;
    for (int s = 0; s < CH; s++) {
        const size_t r = (rowbase + l0 + s) * (size_t)H2;
        float a, u; gate_au(hg[r + h], hg[r + HH + h], a, u);
        A = A * a; U = a * U + u;             // comb((A,U),(a,u))
    }
    Ag[chunk][threadIdx.x] = A; Ug[chunk][threadIdx.x] = U;
    __syncthreads();
    if (chunk == 0) {
        float Pa = 1.f, PuV = 0.f;
        for (int c = 0; c < 8; c++) {
            Pu[c][threadIdx.x] = PuV;         // exclusive prefix (h_{-1} = 0)
            const float na = Ag[c][threadIdx.x], nu = Ug[c][threadIdx.x];
            PuV = na * PuV + nu; Pa *= na;
        }
    }
    __syncthreads();
    float hprev = Pu[chunk][threadIdx.x];
    for (int s = 0; s < CH; s++) {
        const size_t row = rowbase + l0 + s;
        const size_t r = row * (size_t)H2;
        float a, u; gate_au(hg[r + h], hg[r + HH + h], a, u);
        hprev = a * hprev + u;
        hbf[row * HH + h] = (bf16)hprev;
    }
}

// ---------------------------------------------------------------------------
// 4. Output heads: one wave per (b,t); 30 dots of length 512, shuffle-reduced.
// ---------------------------------------------------------------------------
__device__ __forceinline__ float wave_sum(float v) {
#pragma unroll
    for (int off = 16; off > 0; off >>= 1) v += __shfl_xor(v, off, 32);
    return v;
}

__global__ void heads_kernel(const bf16* __restrict__ hbf,
                             const float* __restrict__ Wrtg, const float* __restrict__ brtg,
                             const float* __restrict__ Wmu,  const float* __restrict__ bmu,
                             const float* __restrict__ Wls,  const float* __restrict__ bls,
                             const float* __restrict__ Wst,  const float* __restrict__ bst,
                             float* __restrict__ out) {
    const int bt = blockIdx.x;               // b*T + t
    const int b = bt >> 9, t = bt & 511;
    const int lane = threadIdx.x;
    const bf16* rs = hbf + ((size_t)b * LL + 3 * t) * HH;   // state token
    const bf16* rr = rs + HH;                               // rtg token
    const bf16* ra = rs + 2 * HH;                           // action token

    float artg = 0.f, amu[ADIM] = {}, als[ADIM] = {}, ast[SDIM] = {};
    for (int k = lane; k < HH; k += 32) {
        const float xs = (float)rs[k], xr = (float)rr[k], xa = (float)ra[k];
        artg += xs * Wrtg[k];
#pragma unroll
        for (int o = 0; o < ADIM; o++) { amu[o] += xr * Wmu[k * ADIM + o]; als[o] += xr * Wls[k * ADIM + o]; }
#pragma unroll
        for (int o = 0; o < SDIM; o++) ast[o] += xa * Wst[k * SDIM + o];
    }
    artg = wave_sum(artg);
#pragma unroll
    for (int o = 0; o < ADIM; o++) { amu[o] = wave_sum(amu[o]); als[o] = wave_sum(als[o]); }
#pragma unroll
    for (int o = 0; o < SDIM; o++) ast[o] = wave_sum(ast[o]);

    if (lane == 0) {
        out[bt] = artg + brtg[0];                                   // rtg_preds [B,T,1]
        float* mu = out + BB * TT;                                  // 8192
        float* ls = out + BB * TT + BB * TT * ADIM;                 // 57344
        float* st = out + BB * TT + 2 * BB * TT * ADIM;             // 106496
        for (int o = 0; o < ADIM; o++) {
            mu[bt * ADIM + o] = amu[o] + bmu[o];
            float v = als[o] + bls[o];
            ls[bt * ADIM + o] = fminf(fmaxf(v, -5.f), 2.f);
        }
        for (int o = 0; o < SDIM; o++) st[bt * SDIM + o] = ast[o] + bst[o];
    }
}

// ---------------------------------------------------------------------------
extern "C" void kernel_launch(void* const* d_in, const int* in_sizes, int n_in,
                              void* d_out, int out_size, void* d_ws, size_t ws_size,
                              hipStream_t stream) {
    const int*   timesteps = (const int*)d_in[0];
    const float* states    = (const float*)d_in[1];
    const float* actions   = (const float*)d_in[2];
    const float* rtg       = (const float*)d_in[3];
    const float* emb_t     = (const float*)d_in[4];
    const float* Ws  = (const float*)d_in[5];
    const float* bs  = (const float*)d_in[6];
    const float* Wr  = (const float*)d_in[7];
    const float* br  = (const float*)d_in[8];
    const float* Wa  = (const float*)d_in[9];
    const float* ba  = (const float*)d_in[10];
    const float* lng = (const float*)d_in[11];
    const float* lnb = (const float*)d_in[12];
    const float* gru_W = (const float*)d_in[13];
    const float* gru_b = (const float*)d_in[14];
    const float* Wrtg = (const float*)d_in[15];
    const float* brtg = (const float*)d_in[16];
    const float* Wmu  = (const float*)d_in[17];
    const float* bmu  = (const float*)d_in[18];
    const float* Wls  = (const float*)d_in[19];
    const float* bls  = (const float*)d_in[20];
    const float* Wst  = (const float*)d_in[21];
    const float* bst  = (const float*)d_in[22];

    char* ws = (char*)d_ws;
    bf16*  Wt  = (bf16*)ws;                              // [6][1024][512] bf16
    bf16*  Hbf = (bf16*)(ws + WT_BYTES);                 // [BL][512] bf16
    float* Hg  = (float*)(ws + WT_BYTES + HBF_BYTES);    // [BL][1024] f32

    // 0. weight convert+transpose (all layers once)
    wt_convert_kernel<<<dim3(H2 / 32, HH / 32, NB), dim3(32, 8), 0, stream>>>(gru_W, Wt);

    // 1. fused embedding + projections + LN
    embed_ln_kernel<<<BL, 128, 0, stream>>>(timesteps, states, actions, rtg, emb_t,
                                            Ws, bs, Wr, br, Wa, ba, lng, lnb, Hbf);

    // 2-3. six minGRU blocks: WMMA GEMM then chunked scan
    for (int layer = 0; layer < NB; layer++) {
        gemm_kernel<<<dim3(H2 / 128, BL / 128), 256, 0, stream>>>(
            Hbf, Wt + (size_t)layer * H2 * HH, gru_b + (size_t)layer * H2, Hg);
        scan_kernel<<<BB * 16, dim3(32, 8), 0, stream>>>(Hg, Hbf);
    }

    // 4. output heads
    heads_kernel<<<BB * TT, 32, 0, stream>>>(Hbf, Wrtg, brtg, Wmu, bmu, Wls, bls, Wst, bst,
                                             (float*)d_out);
}
